// CropConvNeuralProcess_58454504898573
// MI455X (gfx1250) — matrix operand order repaired and checked by reading, not compiled, not run on hardware
//
#include <hip/hip_runtime.h>
#include <hip/hip_bf16.h>
#include <stdint.h>

// CropConv (B=32, CIN=COUT=256, 64x64 -> 32x32, k=3 s=2 p=1).
// The reference's crop mask degenerates to all-True (start_idx=1560 >= L=1024),
// so this is a pure strided conv as implicit GEMM:
//   M=256 (cout), N=32768 (b,oh,ow), K=2304 reordered as k' = (kh*3+kw)*256 + ci
// computed with v_wmma_f32_16x16x32_bf16 using a bf16 hi/lo split
// (hi*hi + hi*lo + lo*hi, f32 accumulate -> ~fp32 accuracy).
//
// Weights are repacked once into d_ws as K-reordered bf16 hi/lo planes; the
// per-chunk A tile (128x32 bf16, row stride 2304) is then a perfect 2D tile
// for the Tensor Data Mover (tensor_load_to_lds + s_wait_tensorcnt), issued by
// wave 0 and double-buffered against the WMMA pipe.

typedef __attribute__((ext_vector_type(16))) __bf16 bf16x16;
typedef __attribute__((ext_vector_type(8)))  float  f32x8;

struct __align__(16) Q4 { uint32_t x, y, z, w; };
union BFrag { bf16x16 v; Q4 q[2]; };

__device__ __forceinline__ uint16_t f32_to_bf16_rne(float f) {
    uint32_t u = __builtin_bit_cast(uint32_t, f);
    u += 0x7FFFu + ((u >> 16) & 1u);
    return (uint16_t)(u >> 16);
}
__device__ __forceinline__ float bf16_to_f32(uint16_t h) {
    uint32_t u = (uint32_t)h << 16;
    return __builtin_bit_cast(float, u);
}

#define KTOT   2304
#define KC     32
#define CHUNKS 72            // KTOT / KC
#define WELEMS (256 * KTOT)  // 589824 weight elements

// ---- one-time weight repack: w[co][ci][r] -> wHi/wLo[co][r*256+ci] (bf16) ----
__global__ __launch_bounds__(256)
void repack_weights_kernel(const float* __restrict__ w,
                           uint16_t* __restrict__ wHi,
                           uint16_t* __restrict__ wLo) {
    int i  = blockIdx.x * 256 + threadIdx.x;   // destination index
    int co = i / KTOT;
    int t  = i - co * KTOT;
    int r  = t >> 8;         // kh*3+kw
    int ci = t & 255;
    float f = w[co * KTOT + ci * 9 + r];
    uint16_t hi = f32_to_bf16_rne(f);
    uint16_t lo = f32_to_bf16_rne(f - bf16_to_f32(hi));
    wHi[i] = hi;
    wLo[i] = lo;
}

#if defined(__gfx1250__) && __has_builtin(__builtin_amdgcn_tensor_load_to_lds) && \
    __has_builtin(__builtin_amdgcn_s_wait_tensorcnt)
#define USE_TDM 1
typedef __attribute__((ext_vector_type(4))) unsigned int u32x4;
typedef __attribute__((ext_vector_type(8))) int          i32x8;
typedef __attribute__((ext_vector_type(4))) int          i32x4;

// 2D tile: 128 rows x 32 bf16 cols, row stride 2304 elems -> contiguous 8KB in LDS.
__device__ __forceinline__ void tdm_load_128x32_bf16(const uint16_t* gsrc,
                                                     uint32_t lds_byte_addr) {
    uint64_t ga = (uint64_t)(uintptr_t)gsrc;
    u32x4 g0;
    g0.x = 0x1u;                                            // count=1, user D#
    g0.y = lds_byte_addr;                                   // lds_addr
    g0.z = (uint32_t)ga;                                    // global_addr[31:0]
    g0.w = (uint32_t)((ga >> 32) & 0x1FFFFFFu) | (2u << 30); // addr[56:32] | type=2
    i32x8 g1;
    g1[0] = (1 << 16);                                       // data_size=1 (2 bytes)
    g1[1] = (int)(((unsigned)KTOT & 0xFFFFu) << 16);         // tensor_dim0 lo16
    g1[2] = (int)((((unsigned)KTOT >> 16) & 0xFFFFu) | (256u << 16)); // tdim0 hi | tdim1 lo
    g1[3] = (int)(32u << 16);                                // tdim1 hi=0 | tile_dim0=32
    g1[4] = (int)128u;                                       // tile_dim1=128, tile_dim2=0
    g1[5] = (int)KTOT;                                       // tensor_dim0_stride lo32
    g1[6] = 0;                                               // stride hi | tdim1_stride lo
    g1[7] = 0;
    i32x4 gz4 = {0, 0, 0, 0};                                // groups 2/3 unused (2D)
    i32x8 gz8 = {0, 0, 0, 0, 0, 0, 0, 0};                    // extra group (6-arg form)
    __builtin_amdgcn_tensor_load_to_lds(g0, g1, gz4, gz4, gz8, 0);
}
#endif

// Block: 256 thr = 8 waves, tile M128 x N64. Wave: M16 x N64 (4 subtiles).
__global__ __launch_bounds__(256)
void cropconv_wmma_bf16hilo_kernel(const float* __restrict__ x,
                                   const uint16_t* __restrict__ wHi,
                                   const uint16_t* __restrict__ wLo,
                                   float* __restrict__ out) {
    __shared__ __align__(16) uint16_t sAhi[2][128 * 32];
    __shared__ __align__(16) uint16_t sAlo[2][128 * 32];
    __shared__ __align__(16) uint16_t sBhi[2][64 * 32];
    __shared__ __align__(16) uint16_t sBlo[2][64 * 32];

    const int tid  = threadIdx.x;
    const int lane = tid & 31;
    const int wv   = tid >> 5;
    const int m0   = blockIdx.y * 128;   // cout base
    const int n0   = blockIdx.x * 64;    // pixel base (1024 % 64 == 0 -> single image)

    f32x8 acc[4];
#pragma unroll
    for (int s = 0; s < 4; ++s) acc[s] = 0.0f;

    // ---- loop-invariant im2col geometry: kk = lane, nl = e*8 + wv ----
    int ihb[8], iwb[8], pbase[8];
#pragma unroll
    for (int e = 0; e < 8; ++e) {
        int nl  = e * 8 + wv;
        int n   = n0 + nl;
        int b   = n >> 10;
        int rem = n & 1023;
        int oh  = rem >> 5;
        int ow  = rem & 31;
        ihb[e]  = 2 * oh - 1;
        iwb[e]  = 2 * ow - 1;
        pbase[e] = (b * 256 + lane) * 4096;   // (b*CIN + kk) * H*W
    }

    // A fragment per-lane LDS offsets (ISA 16-bit A 16x32 layout)
    const int aRow  = wv * 16 + (lane & 15);
    const int aOff0 = aRow * 32 + ((lane & 16) ? 8 : 0);
    const int aOff1 = aOff0 + 16;
    // B fragment (bf16 B 32x16): lanes 0-15 K0..7,16..23, lanes 16-31 K8..15,24..31
    const int bCol  = lane & 15;
    const int bKsel = (lane & 16) ? 8 : 0;

#if !defined(USE_TDM)
    Q4 ah0, ah1, al0, al1;
    const int ar0 = tid >> 2,          aq0 = (tid & 3) * 8;
    const int ar1 = (tid + 256) >> 2,  aq1 = aq0;           // (tid+256)&3 == tid&3
#endif

    // ---- staging helpers (inline lambdas) ----
    auto stageA_issue = [&](int kc, int buf) {
#if defined(USE_TDM)
        if (wv == 0) {
            tdm_load_128x32_bf16(wHi + (size_t)m0 * KTOT + kc,
                                 (uint32_t)(uintptr_t)&sAhi[buf][0]);
            tdm_load_128x32_bf16(wLo + (size_t)m0 * KTOT + kc,
                                 (uint32_t)(uintptr_t)&sAlo[buf][0]);
        }
        (void)buf;
#else
        ah0 = *(const Q4*)&wHi[(m0 + ar0) * KTOT + kc + aq0];
        ah1 = *(const Q4*)&wHi[(m0 + ar1) * KTOT + kc + aq1];
        al0 = *(const Q4*)&wLo[(m0 + ar0) * KTOT + kc + aq0];
        al1 = *(const Q4*)&wLo[(m0 + ar1) * KTOT + kc + aq1];
        (void)buf;
#endif
    };
    auto stageA_store = [&](int buf) {
#if !defined(USE_TDM)
        *(Q4*)&sAhi[buf][ar0 * 32 + aq0] = ah0;
        *(Q4*)&sAhi[buf][ar1 * 32 + aq1] = ah1;
        *(Q4*)&sAlo[buf][ar0 * 32 + aq0] = al0;
        *(Q4*)&sAlo[buf][ar1 * 32 + aq1] = al1;
#else
        (void)buf;
#endif
    };
    auto stageB_load = [&](int kc, float* bv) {
        int r   = kc >> 8;                 // kh*3+kw  (scalar per chunk)
        int kh  = r / 3;
        int kw  = r - kh * 3;
        int cof = (kc & 255) * 4096;       // ci0 * H*W
#pragma unroll
        for (int e = 0; e < 8; ++e) {
            int ih = ihb[e] + kh;
            int iw = iwb[e] + kw;
            float f = 0.0f;
            if ((unsigned)ih < 64u && (unsigned)iw < 64u)
                f = x[pbase[e] + cof + ih * 64 + iw];
            bv[e] = f;
        }
    };
    auto stageB_store = [&](const float* bv, int buf) {
#pragma unroll
        for (int e = 0; e < 8; ++e) {
            uint16_t hi = f32_to_bf16_rne(bv[e]);
            uint16_t lo = f32_to_bf16_rne(bv[e] - bf16_to_f32(hi));
            int o = (e * 8 + wv) * 32 + lane;
            sBhi[buf][o] = hi;
            sBlo[buf][o] = lo;
        }
    };
    auto compute = [&](int buf) {
        BFrag ahi, alo;
        ahi.q[0] = *(const Q4*)&sAhi[buf][aOff0];
        ahi.q[1] = *(const Q4*)&sAhi[buf][aOff1];
        alo.q[0] = *(const Q4*)&sAlo[buf][aOff0];
        alo.q[1] = *(const Q4*)&sAlo[buf][aOff1];
#pragma unroll
        for (int s = 0; s < 4; ++s) {
            int bOff = (s * 16 + bCol) * 32 + bKsel;
            BFrag bhi, blo;
            bhi.q[0] = *(const Q4*)&sBhi[buf][bOff];
            bhi.q[1] = *(const Q4*)&sBhi[buf][bOff + 16];
            blo.q[0] = *(const Q4*)&sBlo[buf][bOff];
            blo.q[1] = *(const Q4*)&sBlo[buf][bOff + 16];
            acc[s] = __builtin_amdgcn_wmma_f32_16x16x32_bf16(
                false, ahi.v, false, bhi.v, (short)0, acc[s], false, false);
            acc[s] = __builtin_amdgcn_wmma_f32_16x16x32_bf16(
                false, ahi.v, false, blo.v, (short)0, acc[s], false, false);
            acc[s] = __builtin_amdgcn_wmma_f32_16x16x32_bf16(
                false, alo.v, false, bhi.v, (short)0, acc[s], false, false);
        }
    };

    // ---- prologue: stage chunk 0 into buffer 0 ----
    {
        float bv[8];
        stageA_issue(0, 0);
        stageB_load(0, bv);
        stageA_store(0);
        stageB_store(bv, 0);
#if defined(USE_TDM)
        __builtin_amdgcn_s_wait_tensorcnt(0);
#endif
    }

    // ---- main pipeline: 1 barrier per chunk, double-buffered ----
    for (int c = 0; c < CHUNKS; ++c) {
        __syncthreads();                    // buf[c&1] staged; buf[(c+1)&1] free
        const int cb = c & 1, nb = cb ^ 1;
        float bv[8];
        if (c + 1 < CHUNKS) {
            stageA_issue((c + 1) * KC, nb); // TDM DMA overlaps WMMA below
            stageB_load((c + 1) * KC, bv);  // global gathers in flight
        }
        compute(cb);                        // 12x v_wmma_f32_16x16x32_bf16
        if (c + 1 < CHUNKS) {
            stageA_store(nb);
            stageB_store(bv, nb);
        }
#if defined(USE_TDM)
        __builtin_amdgcn_s_wait_tensorcnt(0);
#endif
    }

    // ---- store: C/D 16x16 f32 layout: VGPR r -> M=r (lanes 0-15) / r+8 (16-31)
    const int coBase = m0 + wv * 16 + ((lane & 16) ? 8 : 0);
#pragma unroll
    for (int s = 0; s < 4; ++s) {
        int n   = n0 + s * 16 + (lane & 15);
        int b   = n >> 10;
        int rem = n & 1023;                 // oh*32 + ow
        float* o = out + (size_t)(b * 256) * 1024 + rem;
#pragma unroll
        for (int r = 0; r < 8; ++r)
            o[(size_t)(coBase + r) * 1024] = acc[s][r];
    }
}

extern "C" void kernel_launch(void* const* d_in, const int* in_sizes, int n_in,
                              void* d_out, int out_size, void* d_ws, size_t ws_size,
                              hipStream_t stream) {
    const float* x = (const float*)d_in[0];   // [32,256,64,64]
    const float* w = (const float*)d_in[1];   // [256,256,3,3]
    float* out = (float*)d_out;               // [32,256,32,32]
    (void)in_sizes; (void)n_in; (void)out_size; (void)ws_size;

    uint16_t* wHi = (uint16_t*)d_ws;          // 589824 bf16
    uint16_t* wLo = wHi + WELEMS;             // 589824 bf16  (2.36 MB total)

    repack_weights_kernel<<<WELEMS / 256, 256, 0, stream>>>(w, wHi, wLo);
    dim3 grid(512, 2, 1);                     // N/64, M/128
    cropconv_wmma_bf16hilo_kernel<<<grid, 256, 0, stream>>>(x, wHi, wLo, out);
}
